// Attention_Critic_29798483100208
// MI455X (gfx1250) — matrix operand order repaired and checked by reading, not compiled, not run on hardware
//
#include <hip/hip_runtime.h>
#include <hip/hip_bf16.h>
#include <cmath>

typedef _Float16 f16;
typedef _Float16 v16h __attribute__((ext_vector_type(16)));
typedef _Float16 v8h  __attribute__((ext_vector_type(8)));
typedef float    v8f  __attribute__((ext_vector_type(8)));

#define B_TOT 32768
#define NAGT  7
#define LP    136   // LDS tile pitch in halves (136*2 bytes = multiple of 16B)

// f16 weight scratch layout (offsets in halves), fragment-major per 16x16x32 tile
#define OFF_W1   0            // 96(pad from 78) x 64
#define OFF_W2   6144         // 64 x 64
#define OFF_W3S  10240        // 64 x 64
#define OFF_WQ   14336        // 64 x 128 (head-concat)
#define OFF_WE   22528        // 7 x (96(pad from 82) x 128)
#define OFF_WK   108544       // 128 x 128 (head-concat)
#define OFF_WV   124928       // 128 x 128 (head-concat)
#define OFF_W3O  141312       // 128 x 64
#define WS_HALVES 149504      // ~299 KB of d_ws

#define INV_SCALE 0.17677669529663687f   // 1/sqrt(32)

__device__ __forceinline__ v8f vzero() {
  v8f v;
#pragma unroll
  for (int i = 0; i < 8; ++i) v[i] = 0.0f;
  return v;
}

__device__ __forceinline__ v8f wmma_f16(v16h a, v16h b, v8f c) {
  // D = A(16x32 f16) x B(32x16 f16) + C(16x16 f32)
  return __builtin_amdgcn_wmma_f32_16x16x32_f16(false, a, false, b, (short)0, c,
                                                false, false);
}

// A-fragment (16x32 f16) from an LDS tile, per ISA 7.12.2 16-bit A layout:
// lane L: M=L%16 ; halves 0..7 -> K = kbase + i + 8*(L/16)
//                  halves 8..15 -> K = kbase + 16 + (i-8) + 8*(L/16)
__device__ __forceinline__ v16h lds_load_A(const f16* tile, int kbase, int lane) {
  int row = lane & 15;
  int g = (lane >> 4) & 1;
  const f16* p = tile + row * LP + kbase + 8 * g;
  union { v16h v; v8h h[2]; } u;
  u.h[0] = *(const v8h*)(p);
  u.h[1] = *(const v8h*)(p + 16);
  return u.v;
}

// B-fragment from fragment-major f16 weight scratch: contiguous 32B per lane.
__device__ __forceinline__ v16h ws_B(const f16* base, int nN, int kc, int nc, int lane) {
  return *(const v16h*)(base + ((((kc * nN + nc) << 5) + lane) << 4));
}

// Store C (f32 accumulator, 16x16 tile) to LDS as f16.
// C layout: lane L holds column n = nbase + L%16, rows r + 8*(L/16).
__device__ __forceinline__ void store_C16(v8f acc, f16* tile, int nbase, int lane) {
  int n = nbase + (lane & 15);
  int g = (lane >> 4) & 1;
#pragma unroll
  for (int r = 0; r < 8; ++r) tile[(r + 8 * g) * LP + n] = (f16)acc[r];
}

// ---------------- weight prep: fp32 -> f16 fragment-major ----------------
// dst element idx = b*(kpad*N) + ((kc*nN + nc)*32 + lane)*16 + i
// B layout: k = kc*32 + (i%8) + 8*(lane/16) + 16*(i/8) ; n = nc*16 + lane%16
__global__ void prep_w(const float* __restrict__ src, f16* __restrict__ dst,
                       int realK, int kpad, int N, int headDim,
                       int batchCount, int srcBatchStride) {
  int idx = blockIdx.x * blockDim.x + threadIdx.x;
  int per = kpad * N;
  int total = batchCount * per;
  if (idx >= total) return;
  int b = idx / per;
  int e = idx - b * per;
  int i = e & 15;
  int lane = (e >> 4) & 31;
  int t = e >> 9;
  int nN = N >> 4;
  int nc = t % nN;
  int kc = t / nN;
  int k = kc * 32 + (i & 7) + 8 * (lane >> 4) + 16 * (i >> 3);
  int n = nc * 16 + (lane & 15);
  float v = 0.0f;
  if (k < realK) {
    if (headDim) {
      int h = n / headDim, d = n - h * headDim;
      v = src[(size_t)b * srcBatchStride + (size_t)h * realK * headDim + (size_t)k * headDim + d];
    } else {
      v = src[(size_t)b * srcBatchStride + (size_t)k * N + n];
    }
  }
  dst[idx] = (f16)v;
}

// ---------------- fused critic kernel ----------------
__global__ __launch_bounds__(256, 1) void critic_fused(
    const float* __restrict__ state_one, const float* __restrict__ act_one,
    const float* __restrict__ state_others, const float* __restrict__ act_others,
    const float* __restrict__ b1, const float* __restrict__ b2,
    const float* __restrict__ be, const float* __restrict__ bv,
    const float* __restrict__ Wout, const float* __restrict__ bout,
    const f16* __restrict__ wf, float* __restrict__ out) {
  __shared__ f16 smem[8][16 * LP];
  const int lane = threadIdx.x & 31;
  const int wid = threadIdx.x >> 5;
  const int m0 = (blockIdx.x * 8 + wid) * 16;
  f16* buf = smem[wid];
  const int l15 = lane & 15;
  const int g = (lane >> 4) & 1;

  // ---------- stage x = [state_one | act_one | 0] (16 x 96) ----------
  for (int idx = lane; idx < 16 * 96; idx += 32) {
    int r = idx / 96, c = idx - (idx / 96) * 96;
    float v = 0.0f;
    if (c < 64) v = state_one[(size_t)(m0 + r) * 64 + c];
    else if (c < 78) v = act_one[(size_t)(m0 + r) * 14 + (c - 64)];
    buf[r * LP + c] = (f16)v;
  }
  __syncthreads();
  v16h ax[3];
#pragma unroll
  for (int kc = 0; kc < 3; ++kc) ax[kc] = lds_load_A(buf, kc * 32, lane);
  __syncthreads();

  // ---------- x1 = relu(x@W1 + b1) ----------
  v8f acc1[4];
#pragma unroll
  for (int nc = 0; nc < 4; ++nc) acc1[nc] = vzero();
#pragma unroll
  for (int kc = 0; kc < 3; ++kc)
#pragma unroll
    for (int nc = 0; nc < 4; ++nc)
      acc1[nc] = wmma_f16(ax[kc], ws_B(wf + OFF_W1, 4, kc, nc, lane), acc1[nc]);
#pragma unroll
  for (int nc = 0; nc < 4; ++nc) {
    float bb = b1[nc * 16 + l15];
#pragma unroll
    for (int r = 0; r < 8; ++r) acc1[nc][r] = fmaxf(acc1[nc][r] + bb, 0.0f);
  }
#pragma unroll
  for (int nc = 0; nc < 4; ++nc) store_C16(acc1[nc], buf, nc * 16, lane);
  __syncthreads();
  v16h a1[2];
  a1[0] = lds_load_A(buf, 0, lane);
  a1[1] = lds_load_A(buf, 32, lane);
  __syncthreads();

  // ---------- sel = x1 @ Wq (head-concat, 16 x 128) ----------
  v8f selC[8];
#pragma unroll
  for (int nc = 0; nc < 8; ++nc) selC[nc] = vzero();
#pragma unroll
  for (int kc = 0; kc < 2; ++kc)
#pragma unroll
    for (int nc = 0; nc < 8; ++nc)
      selC[nc] = wmma_f16(a1[kc], ws_B(wf + OFF_WQ, 8, kc, nc, lane), selC[nc]);

  // ---------- x2 = relu(x1@W2 + b2) ----------
  v8f acc2[4];
#pragma unroll
  for (int nc = 0; nc < 4; ++nc) acc2[nc] = vzero();
#pragma unroll
  for (int kc = 0; kc < 2; ++kc)
#pragma unroll
    for (int nc = 0; nc < 4; ++nc)
      acc2[nc] = wmma_f16(a1[kc], ws_B(wf + OFF_W2, 4, kc, nc, lane), acc2[nc]);
#pragma unroll
  for (int nc = 0; nc < 4; ++nc) {
    float bb = b2[nc * 16 + l15];
#pragma unroll
    for (int r = 0; r < 8; ++r) acc2[nc][r] = fmaxf(acc2[nc][r] + bb, 0.0f);
  }
#pragma unroll
  for (int nc = 0; nc < 4; ++nc) store_C16(acc2[nc], buf, nc * 16, lane);
  __syncthreads();
  v16h a2[2];
  a2[0] = lds_load_A(buf, 0, lane);
  a2[1] = lds_load_A(buf, 32, lane);
  __syncthreads();

  // ---------- x3_self = x2 @ w3_self ----------
  v8f acc3[4];
#pragma unroll
  for (int nc = 0; nc < 4; ++nc) acc3[nc] = vzero();
#pragma unroll
  for (int kc = 0; kc < 2; ++kc)
#pragma unroll
    for (int nc = 0; nc < 4; ++nc)
      acc3[nc] = wmma_f16(a2[kc], ws_B(wf + OFF_W3S, 4, kc, nc, lane), acc3[nc]);

  // ---------- online-softmax attention over agents ----------
  float mxv[4][8], smv[4][8];
  v8f ov[8];
#pragma unroll
  for (int h = 0; h < 4; ++h)
#pragma unroll
    for (int r = 0; r < 8; ++r) { mxv[h][r] = -1e30f; smv[h][r] = 0.0f; }
#pragma unroll
  for (int nc = 0; nc < 8; ++nc) ov[nc] = vzero();

  for (int a = 0; a < NAGT; ++a) {
    // stage inp_a = [state_others[a] | act_others[a] | 0] (16 x 96)
    for (int idx = lane; idx < 16 * 96; idx += 32) {
      int r = idx / 96, c = idx - (idx / 96) * 96;
      float v = 0.0f;
      if (c < 64) v = state_others[((size_t)a * B_TOT + (m0 + r)) * 64 + c];
      else if (c < 82) v = act_others[((size_t)a * B_TOT + (m0 + r)) * 18 + (c - 64)];
      buf[r * LP + c] = (f16)v;
    }
    __syncthreads();
    v16h ai[3];
#pragma unroll
    for (int kc = 0; kc < 3; ++kc) ai[kc] = lds_load_A(buf, kc * 32, lane);
    __syncthreads();

    // enc = relu(inp @ We[a] + be[a])  (16 x 128)
    const f16* we = wf + OFF_WE + (size_t)a * (96 * 128);
    v8f enc[8];
#pragma unroll
    for (int nc = 0; nc < 8; ++nc) enc[nc] = vzero();
#pragma unroll
    for (int kc = 0; kc < 3; ++kc)
#pragma unroll
      for (int nc = 0; nc < 8; ++nc)
        enc[nc] = wmma_f16(ai[kc], ws_B(we, 8, kc, nc, lane), enc[nc]);
#pragma unroll
    for (int nc = 0; nc < 8; ++nc) {
      float bb = be[a * 128 + nc * 16 + l15];
#pragma unroll
      for (int r = 0; r < 8; ++r) enc[nc][r] = fmaxf(enc[nc][r] + bb, 0.0f);
    }
#pragma unroll
    for (int nc = 0; nc < 8; ++nc) store_C16(enc[nc], buf, nc * 16, lane);
    __syncthreads();
    v16h ae[4];
#pragma unroll
    for (int kc = 0; kc < 4; ++kc) ae[kc] = lds_load_A(buf, kc * 32, lane);
    __syncthreads();

    // keys = enc @ Wk (head-concat)
    v8f kk[8];
#pragma unroll
    for (int nc = 0; nc < 8; ++nc) kk[nc] = vzero();
#pragma unroll
    for (int kc = 0; kc < 4; ++kc)
#pragma unroll
      for (int nc = 0; nc < 8; ++nc)
        kk[nc] = wmma_f16(ae[kc], ws_B(wf + OFF_WK, 8, kc, nc, lane), kk[nc]);

    // logits[h] = (sel_h . key_h) / sqrt(d)  -- reduce over the 16 lanes of a group
    float lg[4][8];
#pragma unroll
    for (int h = 0; h < 4; ++h)
#pragma unroll
      for (int r = 0; r < 8; ++r) {
        float p = kk[2 * h][r] * selC[2 * h][r] + kk[2 * h + 1][r] * selC[2 * h + 1][r];
        p += __shfl_xor(p, 1);
        p += __shfl_xor(p, 2);
        p += __shfl_xor(p, 4);
        p += __shfl_xor(p, 8);
        lg[h][r] = p * INV_SCALE;
      }

    // vals = relu(enc @ Wv + bv) (reuse kk)
#pragma unroll
    for (int nc = 0; nc < 8; ++nc) kk[nc] = vzero();
#pragma unroll
    for (int kc = 0; kc < 4; ++kc)
#pragma unroll
      for (int nc = 0; nc < 8; ++nc)
        kk[nc] = wmma_f16(ae[kc], ws_B(wf + OFF_WV, 8, kc, nc, lane), kk[nc]);
#pragma unroll
    for (int nc = 0; nc < 8; ++nc) {
      float bb = bv[nc * 16 + l15];
#pragma unroll
      for (int r = 0; r < 8; ++r) kk[nc][r] = fmaxf(kk[nc][r] + bb, 0.0f);
    }

    // online softmax update
#pragma unroll
    for (int h = 0; h < 4; ++h)
#pragma unroll
      for (int r = 0; r < 8; ++r) {
        float l = lg[h][r];
        float mo = mxv[h][r];
        float mn = fmaxf(mo, l);
        float corr = __expf(mo - mn);
        float w = __expf(l - mn);
        smv[h][r] = smv[h][r] * corr + w;
        ov[2 * h][r] = ov[2 * h][r] * corr + w * kk[2 * h][r];
        ov[2 * h + 1][r] = ov[2 * h + 1][r] * corr + w * kk[2 * h + 1][r];
        mxv[h][r] = mn;
      }
  }

  // normalize -> x2_others (16 x 128)
#pragma unroll
  for (int nc = 0; nc < 8; ++nc) {
    int h = nc >> 1;
#pragma unroll
    for (int r = 0; r < 8; ++r) ov[nc][r] = ov[nc][r] / smv[h][r];
  }
  __syncthreads();
#pragma unroll
  for (int nc = 0; nc < 8; ++nc) store_C16(ov[nc], buf, nc * 16, lane);
  __syncthreads();
  v16h ao[4];
#pragma unroll
  for (int kc = 0; kc < 4; ++kc) ao[kc] = lds_load_A(buf, kc * 32, lane);

  // x3 = relu(x3_self + x2_others @ w3_others)
  v8f xf[4];
#pragma unroll
  for (int nc = 0; nc < 4; ++nc) xf[nc] = acc3[nc];
#pragma unroll
  for (int kc = 0; kc < 4; ++kc)
#pragma unroll
    for (int nc = 0; nc < 4; ++nc)
      xf[nc] = wmma_f16(ao[kc], ws_B(wf + OFF_W3O, 4, kc, nc, lane), xf[nc]);
#pragma unroll
  for (int nc = 0; nc < 4; ++nc)
#pragma unroll
    for (int r = 0; r < 8; ++r) xf[nc][r] = fmaxf(xf[nc][r], 0.0f);

  // out = x3 @ Wout + bout  (per-row dot, lane reduce within 16-lane group)
  float wv0 = Wout[0 * 16 + l15];
  float wv1 = Wout[1 * 16 + l15];
  float wv2 = Wout[2 * 16 + l15];
  float wv3 = Wout[3 * 16 + l15];
  float bo = bout[0];
#pragma unroll
  for (int r = 0; r < 8; ++r) {
    float t = xf[0][r] * wv0 + xf[1][r] * wv1 + xf[2][r] * wv2 + xf[3][r] * wv3;
    t += __shfl_xor(t, 1);
    t += __shfl_xor(t, 2);
    t += __shfl_xor(t, 4);
    t += __shfl_xor(t, 8);
    if (l15 == 0) out[m0 + r + 8 * g] = t + bo;
  }
}

extern "C" void kernel_launch(void* const* d_in, const int* in_sizes, int n_in,
                              void* d_out, int out_size, void* d_ws, size_t ws_size,
                              hipStream_t stream) {
  (void)in_sizes; (void)n_in; (void)out_size; (void)ws_size;
  const float* state_one    = (const float*)d_in[0];
  const float* act_one      = (const float*)d_in[1];
  const float* state_others = (const float*)d_in[2];
  const float* act_others   = (const float*)d_in[3];
  const float* W1           = (const float*)d_in[4];
  const float* b1           = (const float*)d_in[5];
  const float* W2           = (const float*)d_in[6];
  const float* b2           = (const float*)d_in[7];
  const float* w3_self      = (const float*)d_in[8];
  const float* We           = (const float*)d_in[9];
  const float* be           = (const float*)d_in[10];
  const float* Wk           = (const float*)d_in[11];
  const float* Wq           = (const float*)d_in[12];
  const float* Wv           = (const float*)d_in[13];
  const float* bv           = (const float*)d_in[14];
  const float* w3_others    = (const float*)d_in[15];
  const float* Wout         = (const float*)d_in[16];
  const float* bout         = (const float*)d_in[17];
  float* out = (float*)d_out;
  f16* wf = (f16*)d_ws;

  auto launch_prep = [&](const float* src, long off, int realK, int kpad, int N,
                         int hd, int batch, int sstride) {
    int total = batch * kpad * N;
    prep_w<<<(total + 255) / 256, 256, 0, stream>>>(src, wf + off, realK, kpad, N,
                                                    hd, batch, sstride);
  };
  launch_prep(W1,        OFF_W1,  78, 96, 64,  0,  1, 0);
  launch_prep(W2,        OFF_W2,  64, 64, 64,  0,  1, 0);
  launch_prep(w3_self,   OFF_W3S, 64, 64, 64,  0,  1, 0);
  launch_prep(Wq,        OFF_WQ,  64, 64, 128, 32, 1, 0);
  launch_prep(We,        OFF_WE,  82, 96, 128, 0,  7, 82 * 128);
  launch_prep(Wk,        OFF_WK, 128, 128, 128, 32, 1, 0);
  launch_prep(Wv,        OFF_WV, 128, 128, 128, 32, 1, 0);
  launch_prep(w3_others, OFF_W3O, 128, 128, 64, 0,  1, 0);

  // 2048 row-tiles of 16, 8 waves (tiles) per 256-thread block -> 256 blocks
  critic_fused<<<dim3(B_TOT / 128), dim3(256), 0, stream>>>(
      state_one, act_one, state_others, act_others,
      b1, b2, be, bv, Wout, bout, wf, out);
}
